// MultiHeadAttentionRelative_11055245820012
// MI455X (gfx1250) — matrix operand order repaired and checked by reading, not compiled, not run on hardware
//
#include <hip/hip_runtime.h>

// ---------------------------------------------------------------------------
// MHA with bidirectional ALiBi, bf16 WMMA + TDM pipeline for gfx1250 (wave32).
// B=4, S=2048, H=16, D=64, MODEL=1024.
// ---------------------------------------------------------------------------

#define MODEL_DIM 1024
#define SEQ       2048
#define BATCH     4
#define HEADS     16
#define DHEAD     64

typedef __attribute__((ext_vector_type(16))) __bf16        v16bf;
typedef __attribute__((ext_vector_type(8)))  float         v8f;
typedef __attribute__((ext_vector_type(4)))  unsigned int  v4u;
typedef __attribute__((ext_vector_type(8)))  int           v8i;
typedef __attribute__((ext_vector_type(4)))  int           v4i;

// f32 -> bf16 (round-to-nearest-even)
static __device__ inline unsigned short bf16_bits(float f) {
    union { float f; unsigned u; } v; v.f = f;
    unsigned r = v.u + 0x7FFFu + ((v.u >> 16) & 1u);
    return (unsigned short)(r >> 16);
}

// Build a 16-element bf16 fragment from two 16-byte chunks.
static __device__ inline v16bf make_frag(const unsigned short* p0, const unsigned short* p1) {
    union { uint4 q[2]; v16bf v; } u;
    __builtin_memcpy(&u.q[0], p0, 16);
    __builtin_memcpy(&u.q[1], p1, 16);
    return u.v;
}

// A-matrix fragment (16x32, ISA 7.12.2): lane = row, half h holds
// K = h*8 + {0..7} and 16 + h*8 + {0..7}.
static __device__ inline v16bf frag_a(const unsigned short* base, int stride) {
    const int lane = threadIdx.x & 31;
    const int r = lane & 15, h = (lane >> 4) & 1;
    const unsigned short* p = base + r * stride + h * 8;
    return make_frag(p, p + 16);
}

// B-matrix fragment (32x16): lane = column n, half h holds
// K = h*8 + {0..7} and 16 + h*8 + {0..7}; tile stored [n rows][contraction contiguous].
static __device__ inline v16bf frag_b(const unsigned short* base, int stride) {
    const int lane = threadIdx.x & 31;
    const int n = lane & 15, h = (lane >> 4) & 1;
    const unsigned short* p = base + n * stride + h * 8;
    return make_frag(p, p + 16);
}

static __device__ inline v8f wmma_bf16(v16bf a, v16bf b, v8f c) {
    return __builtin_amdgcn_wmma_f32_16x16x32_bf16(
        false, a, false, b, (short)0, c, false, false);
}

// 16-lane (half-wave) reductions; xor masks < 16 stay within a half.
static __device__ inline float rmax16(float v) {
#pragma unroll
    for (int m = 1; m < 16; m <<= 1) v = fmaxf(v, __shfl_xor(v, m, 32));
    return v;
}
static __device__ inline float rsum16(float v) {
#pragma unroll
    for (int m = 1; m < 16; m <<= 1) v += __shfl_xor(v, m, 32);
    return v;
}

// ---------------------------------------------------------------------------
// TDM: 2D bf16 tile load, global -> LDS, with LDS row padding.
// D# packing per CDNA5 ISA ch.8 (group0 128b, group1 256b; groups 2/3 zero
// for 2D tiles). pad_interval code v -> pad after 2^(v+1) DWORDs; pad_amount
// code a -> (a+1) DWORDs inserted. 6-arg builtin (this toolchain).
// ---------------------------------------------------------------------------
static __device__ inline void tdm_load_2d(
    unsigned lds_addr, const unsigned short* gptr,
    unsigned tensor_d0, unsigned tensor_d1,   // tensor extents (elements)
    unsigned tile_d0,   unsigned tile_d1,     // tile extents (elements)
    unsigned long long stride0,               // row stride (elements)
    unsigned pad_interval_code, unsigned pad_amount_code)
{
    unsigned long long ga = (unsigned long long)(size_t)gptr;
    v4u g0;
    g0[0] = 1u;                                     // count=1, user mode
    g0[1] = lds_addr;                               // LDS byte address
    g0[2] = (unsigned)(ga & 0xFFFFFFFFu);           // global addr [31:0]
    g0[3] = (unsigned)((ga >> 32) & 0x01FFFFFFu)    // global addr [56:32]
          | (2u << 30);                             // type = 2 ("image")
    v8i g1;
    unsigned flags = (1u << 16)                     // data_size = 2 bytes
                   | (1u << 20)                     // pad_enable
                   | (pad_interval_code << 22)
                   | (pad_amount_code   << 25);
    g1[0] = (int)flags;                             // wg_mask=0 (not in cluster)
    g1[1] = (int)((tensor_d0 & 0xFFFFu) << 16);     // abar=0 | tensor_dim0 lo16
    g1[2] = (int)((tensor_d0 >> 16) | ((tensor_d1 & 0xFFFFu) << 16));
    g1[3] = (int)((tensor_d1 >> 16) | (tile_d0 << 16));
    g1[4] = (int)(tile_d1 & 0xFFFFu);               // tile_dim1 | tile_dim2=0
    g1[5] = (int)(unsigned)(stride0 & 0xFFFFFFFFu); // tensor_dim0_stride lo32
    g1[6] = (int)(unsigned)((stride0 >> 32) & 0xFFFFu);
    g1[7] = 0;
    v4i z4 = {0, 0, 0, 0};
    v8i z8 = {0, 0, 0, 0, 0, 0, 0, 0};
    __builtin_amdgcn_tensor_load_to_lds(g0, g1, z4, z4, z8, 0);
}

static __device__ inline unsigned lds_addr_of(const void* p) {
    return (unsigned)(size_t)p;   // LDS aperture: low 32 bits = LDS byte address
}

// ---------------------------------------------------------------------------
// Kernel 0: f32 -> bf16 conversion (vectorized, n multiple of 4).
// ---------------------------------------------------------------------------
__global__ __launch_bounds__(256) void cvt_bf16_kernel(
    const float* __restrict__ src, unsigned short* __restrict__ dst, int n)
{
    int i = (blockIdx.x * 256 + threadIdx.x) * 4;
    if (i < n) {
        float4 f = *reinterpret_cast<const float4*>(src + i);
        union { unsigned short u[4]; uint2 v; } o;
        o.u[0] = bf16_bits(f.x); o.u[1] = bf16_bits(f.y);
        o.u[2] = bf16_bits(f.z); o.u[3] = bf16_bits(f.w);
        *reinterpret_cast<uint2*>(dst + i) = o.v;
    }
}

// ---------------------------------------------------------------------------
// Kernel 1: fused Q/K/V projection, TDM double-buffered GEMM.
// Y = X @ W^T + b. Block tile 128x128, 8 waves of 32x64 (8 WMMA / K-step).
// blockIdx.z selects {Q,K,V}. Q,K -> [B,H,S,D] bf16; V -> [B,H,D,S] bf16.
// ---------------------------------------------------------------------------
__global__ __launch_bounds__(256) void qkv_proj_kernel(
    const unsigned short* __restrict__ Xqb, const unsigned short* __restrict__ Xkb,
    const unsigned short* __restrict__ Xvb,
    const unsigned short* __restrict__ Wqb, const unsigned short* __restrict__ Wkb,
    const unsigned short* __restrict__ Wvb,
    const float* __restrict__ bq, const float* __restrict__ bk, const float* __restrict__ bv,
    unsigned short* __restrict__ Qbf, unsigned short* __restrict__ Kbf,
    unsigned short* __restrict__ Vt)
{
    __shared__ unsigned short As[2][128 * 40];
    __shared__ unsigned short Bs[2][128 * 40];

    const int which = blockIdx.z;
    const unsigned short* X;  const unsigned short* W;  const float* bias;
    unsigned short* Out;
    if (which == 0)      { X = Xqb; W = Wqb; bias = bq; Out = Qbf; }
    else if (which == 1) { X = Xkb; W = Wkb; bias = bk; Out = Kbf; }
    else                 { X = Xvb; W = Wvb; bias = bv; Out = Vt;  }

    const int t = threadIdx.x, lane = t & 31, wid = t >> 5;
    const int wm = wid & 3, wn = wid >> 2;          // 4x2 waves, 32x64 each
    const int m0 = blockIdx.x * 128;
    const int n0 = blockIdx.y * 128;

    v8f acc[2][4] = {};

    // Prime buffer 0. Tiles: A = [128 rows][32 k], B = [128 rows][32 k],
    // LDS row stride 40 elems (pad 16DW interval -> code 3, 4DW amount -> code 3).
    if (wid == 0) {
        tdm_load_2d(lds_addr_of(&As[0][0]), X + (size_t)m0 * MODEL_DIM,
                    MODEL_DIM, 8192, 32, 128, MODEL_DIM, 3, 3);
        tdm_load_2d(lds_addr_of(&Bs[0][0]), W + (size_t)n0 * MODEL_DIM,
                    MODEL_DIM, MODEL_DIM, 32, 128, MODEL_DIM, 3, 3);
    }

    int cur = 0;
    for (int kk = 0; kk < MODEL_DIM; kk += 32) {
        if (wid == 0) {
            if (kk + 32 < MODEL_DIM) {
                tdm_load_2d(lds_addr_of(&As[cur ^ 1][0]),
                            X + (size_t)m0 * MODEL_DIM + kk + 32,
                            MODEL_DIM, 8192, 32, 128, MODEL_DIM, 3, 3);
                tdm_load_2d(lds_addr_of(&Bs[cur ^ 1][0]),
                            W + (size_t)n0 * MODEL_DIM + kk + 32,
                            MODEL_DIM, MODEL_DIM, 32, 128, MODEL_DIM, 3, 3);
                __builtin_amdgcn_s_wait_tensorcnt(2);   // current buffer done
            } else {
                __builtin_amdgcn_s_wait_tensorcnt(0);
            }
        }
        __syncthreads();

        v16bf a0 = frag_a(&As[cur][0] + (wm * 32 +  0) * 40, 40);
        v16bf a1 = frag_a(&As[cur][0] + (wm * 32 + 16) * 40, 40);
#pragma unroll
        for (int ni = 0; ni < 4; ++ni) {
            v16bf bn = frag_b(&Bs[cur][0] + (wn * 64 + ni * 16) * 40, 40);
            acc[0][ni] = wmma_bf16(a0, bn, acc[0][ni]);
            acc[1][ni] = wmma_bf16(a1, bn, acc[1][ni]);
        }
        __syncthreads();
        cur ^= 1;
    }

    // C/D layout: lane holds column c = lane&15, rows g + 8*h.
    const int h = (lane >> 4) & 1, c = lane & 15;
#pragma unroll
    for (int mi = 0; mi < 2; ++mi)
#pragma unroll
        for (int ni = 0; ni < 4; ++ni)
#pragma unroll
            for (int g = 0; g < 8; ++g) {
                int m = m0 + wm * 32 + mi * 16 + g + 8 * h;
                int n = n0 + wn * 64 + ni * 16 + c;
                float val = acc[mi][ni][g] + bias[n];
                int b = m >> 11, s = m & (SEQ - 1);
                int head = n >> 6, d = n & 63;
                size_t addr;
                if (which == 2)  // V transposed: [B,H,D,S]
                    addr = ((size_t)(b * HEADS + head) * DHEAD + d) * SEQ + s;
                else             // Q,K: [B,H,S,D]
                    addr = ((size_t)(b * HEADS + head) * SEQ + s) * DHEAD + d;
                Out[addr] = bf16_bits(val);
            }
}

// ---------------------------------------------------------------------------
// Kernel 2: flash attention with bidirectional ALiBi, TDM-fed K/V tiles.
// Grid (S/128, HEADS, BATCH); wave owns 16 query rows; 8 WMMA / 32-key step.
// ---------------------------------------------------------------------------
__global__ __launch_bounds__(256) void attn_kernel(
    const unsigned short* __restrict__ Qbf, const unsigned short* __restrict__ Kbf,
    const unsigned short* __restrict__ Vt,  unsigned short* __restrict__ Obf)
{
    __shared__ unsigned short Ks[2][32 * 72];    // [k rows][d contiguous], pad->72
    __shared__ unsigned short Vs[2][64 * 40];    // [d rows][k contiguous], pad->40
    __shared__ unsigned short Ps[8][16 * 40];    // per-wave P tile 16x32

    const int t = threadIdx.x, lane = t & 31, wid = t >> 5;
    const int h_ = (lane >> 4) & 1, c = lane & 15;
    const int head = blockIdx.y, b = blockIdx.z;
    const int q0 = blockIdx.x * 128 + wid * 16;

    const size_t bh = (size_t)(b * HEADS + head);
    const unsigned short* Qrow  = Qbf + (bh * SEQ + q0) * DHEAD;
    const unsigned short* Kbase = Kbf + bh * SEQ * DHEAD;
    const unsigned short* Vbase = Vt  + bh * DHEAD * SEQ;

    v16bf qf0 = frag_a(Qrow,      DHEAD);   // d 0..31
    v16bf qf1 = frag_a(Qrow + 32, DHEAD);   // d 32..63

    const float slope = __expf(-(float)((head & 7) + 1) * 0.39725673f); // 24^(-(i+1)/8)
    const bool  fwd   = head < 8;

    v8f o[4] = {};
    float mrun[8], lrun[8];
#pragma unroll
    for (int g = 0; g < 8; ++g) { mrun[g] = -3.0e38f; lrun[g] = 0.f; }

    // Prime buffer 0: K tile 32x64 (row 32DW -> interval code 4, pad 4DW -> 3),
    // V^T tile 64x32 (row 16DW -> interval code 3, pad 4DW -> 3).
    if (wid == 0) {
        tdm_load_2d(lds_addr_of(&Ks[0][0]), Kbase, DHEAD, SEQ, 64, 32, DHEAD, 4, 3);
        tdm_load_2d(lds_addr_of(&Vs[0][0]), Vbase, SEQ, DHEAD, 32, 64, SEQ,   3, 3);
    }

    int cur = 0;
    for (int kb = 0; kb < SEQ; kb += 32) {
        if (wid == 0) {
            if (kb + 32 < SEQ) {
                tdm_load_2d(lds_addr_of(&Ks[cur ^ 1][0]),
                            Kbase + (size_t)(kb + 32) * DHEAD,
                            DHEAD, SEQ, 64, 32, DHEAD, 4, 3);
                tdm_load_2d(lds_addr_of(&Vs[cur ^ 1][0]),
                            Vbase + kb + 32,
                            SEQ, DHEAD, 32, 64, SEQ, 3, 3);
                __builtin_amdgcn_s_wait_tensorcnt(2);
            } else {
                __builtin_amdgcn_s_wait_tensorcnt(0);
            }
        }
        __syncthreads();

        // Scores: two 16x16 column tiles, contraction over d = 2x32.
        v8f s[2];
#pragma unroll
        for (int kt = 0; kt < 2; ++kt) {
            v16bf bk0 = frag_b(&Ks[cur][0] + (kt * 16) * 72,      72);
            v16bf bk1 = frag_b(&Ks[cur][0] + (kt * 16) * 72 + 32, 72);
            v8f z = {};
            z = wmma_bf16(qf0, bk0, z);
            z = wmma_bf16(qf1, bk1, z);
            s[kt] = z;
        }

        // Scale + ALiBi + online softmax.
        float pv[2][8], mt[8];
#pragma unroll
        for (int g = 0; g < 8; ++g) {
            const int i = q0 + g + 8 * h_;
            float rowmax = -3.0e38f;
#pragma unroll
            for (int kt = 0; kt < 2; ++kt) {
                int j = kb + kt * 16 + c;
                float bias = fwd ? ((j <= i) ? slope * (float)(j - i) : -1.0e9f)
                                 : ((j >= i) ? slope * (float)(i - j) : -1.0e9f);
                float sv = s[kt][g] * 0.125f + bias;   // 1/sqrt(64)
                pv[kt][g] = sv;
                rowmax = fmaxf(rowmax, sv);
            }
            mt[g] = rmax16(rowmax);
        }
#pragma unroll
        for (int g = 0; g < 8; ++g) {
            float mnew  = fmaxf(mrun[g], mt[g]);
            float alpha = __expf(mrun[g] - mnew);
            float rsum  = 0.f;
#pragma unroll
            for (int kt = 0; kt < 2; ++kt) {
                float p = __expf(pv[kt][g] - mnew);
                pv[kt][g] = p;
                rsum += p;
            }
            rsum = rsum16(rsum);
            lrun[g] = lrun[g] * alpha + rsum;
            mrun[g] = mnew;
#pragma unroll
            for (int dt = 0; dt < 4; ++dt) o[dt][g] *= alpha;
        }

        // P (C/D layout) -> bf16 A-fragment via per-wave LDS tile.
#pragma unroll
        for (int g = 0; g < 8; ++g)
#pragma unroll
            for (int kt = 0; kt < 2; ++kt)
                Ps[wid][(g + 8 * h_) * 40 + kt * 16 + c] = bf16_bits(pv[kt][g]);

        v16bf pf = frag_a(&Ps[wid][0], 40);
#pragma unroll
        for (int dt = 0; dt < 4; ++dt) {
            v16bf bv_ = frag_b(&Vs[cur][0] + (dt * 16) * 40, 40);
            o[dt] = wmma_bf16(pf, bv_, o[dt]);
        }
        __syncthreads();
        cur ^= 1;
    }

    // Normalize; write O token-major [B,S,MODEL] bf16 for the out-proj GEMM.
#pragma unroll
    for (int g = 0; g < 8; ++g) {
        float inv = 1.0f / lrun[g];
        int srow = q0 + g + 8 * h_;
#pragma unroll
        for (int dt = 0; dt < 4; ++dt) {
            size_t addr = ((size_t)b * SEQ + srow) * MODEL_DIM + head * DHEAD + dt * 16 + c;
            Obf[addr] = bf16_bits(o[dt][g] * inv);
        }
    }
}

// ---------------------------------------------------------------------------
// Kernel 3: output projection Y = O @ Wo^T + bo (bf16 in, f32 out), TDM-fed.
// ---------------------------------------------------------------------------
__global__ __launch_bounds__(256) void out_proj_kernel(
    const unsigned short* __restrict__ Obf, const unsigned short* __restrict__ Wob,
    const float* __restrict__ bo, float* __restrict__ Y)
{
    __shared__ unsigned short As[2][128 * 40];
    __shared__ unsigned short Bs[2][128 * 40];

    const int t = threadIdx.x, lane = t & 31, wid = t >> 5;
    const int wm = wid & 3, wn = wid >> 2;
    const int m0 = blockIdx.x * 128;
    const int n0 = blockIdx.y * 128;

    v8f acc[2][4] = {};

    if (wid == 0) {
        tdm_load_2d(lds_addr_of(&As[0][0]), Obf + (size_t)m0 * MODEL_DIM,
                    MODEL_DIM, 8192, 32, 128, MODEL_DIM, 3, 3);
        tdm_load_2d(lds_addr_of(&Bs[0][0]), Wob + (size_t)n0 * MODEL_DIM,
                    MODEL_DIM, MODEL_DIM, 32, 128, MODEL_DIM, 3, 3);
    }

    int cur = 0;
    for (int kk = 0; kk < MODEL_DIM; kk += 32) {
        if (wid == 0) {
            if (kk + 32 < MODEL_DIM) {
                tdm_load_2d(lds_addr_of(&As[cur ^ 1][0]),
                            Obf + (size_t)m0 * MODEL_DIM + kk + 32,
                            MODEL_DIM, 8192, 32, 128, MODEL_DIM, 3, 3);
                tdm_load_2d(lds_addr_of(&Bs[cur ^ 1][0]),
                            Wob + (size_t)n0 * MODEL_DIM + kk + 32,
                            MODEL_DIM, MODEL_DIM, 32, 128, MODEL_DIM, 3, 3);
                __builtin_amdgcn_s_wait_tensorcnt(2);
            } else {
                __builtin_amdgcn_s_wait_tensorcnt(0);
            }
        }
        __syncthreads();

        v16bf a0 = frag_a(&As[cur][0] + (wm * 32 +  0) * 40, 40);
        v16bf a1 = frag_a(&As[cur][0] + (wm * 32 + 16) * 40, 40);
#pragma unroll
        for (int ni = 0; ni < 4; ++ni) {
            v16bf bn = frag_b(&Bs[cur][0] + (wn * 64 + ni * 16) * 40, 40);
            acc[0][ni] = wmma_bf16(a0, bn, acc[0][ni]);
            acc[1][ni] = wmma_bf16(a1, bn, acc[1][ni]);
        }
        __syncthreads();
        cur ^= 1;
    }

    const int h = (lane >> 4) & 1, c = lane & 15;
#pragma unroll
    for (int mi = 0; mi < 2; ++mi)
#pragma unroll
        for (int ni = 0; ni < 4; ++ni)
#pragma unroll
            for (int g = 0; g < 8; ++g) {
                int m = m0 + wm * 32 + mi * 16 + g + 8 * h;
                int n = n0 + wn * 64 + ni * 16 + c;
                Y[(size_t)m * MODEL_DIM + n] = acc[mi][ni][g] + bo[n];
            }
}

// ---------------------------------------------------------------------------
extern "C" void kernel_launch(void* const* d_in, const int* in_sizes, int n_in,
                              void* d_out, int out_size, void* d_ws, size_t ws_size,
                              hipStream_t stream) {
    (void)in_sizes; (void)n_in; (void)out_size; (void)ws_size;
    const float* query = (const float*)d_in[0];
    const float* key   = (const float*)d_in[1];
    const float* value = (const float*)d_in[2];
    const float* Wq = (const float*)d_in[3];
    const float* bq = (const float*)d_in[4];
    const float* Wk = (const float*)d_in[5];
    const float* bk = (const float*)d_in[6];
    const float* Wv = (const float*)d_in[7];
    const float* bv = (const float*)d_in[8];
    const float* Wo = (const float*)d_in[9];
    const float* bo = (const float*)d_in[10];

    const size_t MB = 1024 * 1024;
    char* ws = (char*)d_ws;
    unsigned short* Xqb = (unsigned short*)(ws);                  // 16 MB each
    unsigned short* Xkb = (unsigned short*)(ws + 16 * MB);
    unsigned short* Xvb = (unsigned short*)(ws + 32 * MB);
    unsigned short* Qbf = (unsigned short*)(ws + 48 * MB);
    unsigned short* Kbf = (unsigned short*)(ws + 64 * MB);
    unsigned short* Vt  = (unsigned short*)(ws + 80 * MB);
    unsigned short* Obf = (unsigned short*)(ws + 96 * MB);
    unsigned short* Wqb = (unsigned short*)(ws + 112 * MB);       // 2 MB each
    unsigned short* Wkb = (unsigned short*)(ws + 114 * MB);
    unsigned short* Wvb = (unsigned short*)(ws + 116 * MB);
    unsigned short* Wob = (unsigned short*)(ws + 118 * MB);

    dim3 blk(256, 1, 1);
    const int NX = BATCH * SEQ * MODEL_DIM;        // 8388608
    const int NW = MODEL_DIM * MODEL_DIM;          // 1048576
    cvt_bf16_kernel<<<NX / 1024, blk, 0, stream>>>(query, Xqb, NX);
    cvt_bf16_kernel<<<NX / 1024, blk, 0, stream>>>(key,   Xkb, NX);
    cvt_bf16_kernel<<<NX / 1024, blk, 0, stream>>>(value, Xvb, NX);
    cvt_bf16_kernel<<<NW / 1024, blk, 0, stream>>>(Wq, Wqb, NW);
    cvt_bf16_kernel<<<NW / 1024, blk, 0, stream>>>(Wk, Wkb, NW);
    cvt_bf16_kernel<<<NW / 1024, blk, 0, stream>>>(Wv, Wvb, NW);
    cvt_bf16_kernel<<<NW / 1024, blk, 0, stream>>>(Wo, Wob, NW);

    qkv_proj_kernel<<<dim3(64, 8, 3), blk, 0, stream>>>(
        Xqb, Xkb, Xvb, Wqb, Wkb, Wvb, bq, bk, bv, Qbf, Kbf, Vt);
    attn_kernel<<<dim3(16, 16, 4), blk, 0, stream>>>(Qbf, Kbf, Vt, Obf);
    out_proj_kernel<<<dim3(64, 8, 1), blk, 0, stream>>>(Obf, Wob, bo, (float*)d_out);
}
